// GGMLLayer_15745350107702
// MI455X (gfx1250) — matrix operand order repaired and checked by reading, not compiled, not run on hardware
//
#include <hip/hip_runtime.h>

// ---------------------------------------------------------------------------
// GGML Q4-style dequant + LoRA patch + linear layer for MI455X (gfx1250).
//   W = (q - 8) * scale + lora_a @ lora_b            [O, I]  (materialized bf16)
//   y = x @ W^T + bias                               [T, O]  (bf16 WMMA, f32 acc)
// GEMM: manual ping-pong double buffering (no rotation copies), immediate-
// offset addressing (K hardcoded), 32Mx64N per wave, 64Mx256N per block.
// ---------------------------------------------------------------------------

typedef __attribute__((ext_vector_type(16))) __bf16 v16bf;
typedef __attribute__((ext_vector_type(8)))  __bf16 v8bf;
typedef __attribute__((ext_vector_type(8)))  float  v8f;

constexpr int KDIM = 4096;   // I (reduction dim) — fixed by the harness shapes

__device__ __forceinline__ unsigned short f32_to_bf16_raw(float f) {
  union { float f; unsigned int u; } v; v.f = f;
  unsigned int rnd = 0x7FFFu + ((v.u >> 16) & 1u);   // round-to-nearest-even
  return (unsigned short)((v.u + rnd) >> 16);
}

// ---- Kernel 1: W bf16 = dequant(q, scales) + lora_a @ lora_b --------------
__global__ __launch_bounds__(256) void ggml_prep_w(
    const int* __restrict__ qw, const float* __restrict__ scales,
    const float* __restrict__ la, const float* __restrict__ lb,
    unsigned short* __restrict__ Wb, int O, int I, int R) {
  long long idx = (long long)blockIdx.x * blockDim.x + threadIdx.x;
  long long total = (long long)O * I;
  if (idx >= total) return;
  int o = (int)(idx / I);
  int i = (int)(idx % I);
  float w = ((float)qw[idx] - 8.0f) * scales[(long long)o * (I >> 5) + (i >> 5)];
  for (int r = 0; r < R; ++r)
    w += la[(long long)o * R + r] * lb[(long long)r * I + i];
  Wb[idx] = f32_to_bf16_raw(w);
}

// ---- Kernel 2: x f32 -> bf16 ----------------------------------------------
__global__ __launch_bounds__(256) void ggml_prep_x(
    const float* __restrict__ x, unsigned short* __restrict__ xb, long long n) {
  long long idx = (long long)blockIdx.x * blockDim.x + threadIdx.x;
  if (idx < n) xb[idx] = f32_to_bf16_raw(x[idx]);
}

// ---- Fragment group load (immediate offsets off pa/pb) --------------------
// pa = X row (M0+lr) base + half*16  : A 16x32 map, two 16B chunks per frag
// pb = W row (N0+lr) base + half*16  : B 32x16 map, one 32B chunk per tile
template <int OFFE>  // element offset of this K-step from pa/pb (0 or 32)
__device__ __forceinline__ void load_set(const __bf16* __restrict__ pa,
                                         const __bf16* __restrict__ pb,
                                         v16bf& a0, v16bf& a1, v16bf (&b)[4]) {
  constexpr long long A1 = 16LL * KDIM;  // 16 rows down (bytes: 128KB, fits imm24)
  v8bf c0 = *(const v8bf*)(pa + OFFE);
  v8bf c1 = *(const v8bf*)(pa + OFFE + 8);
  v8bf d0 = *(const v8bf*)(pa + A1 + OFFE);
  v8bf d1 = *(const v8bf*)(pa + A1 + OFFE + 8);
#pragma unroll
  for (int i = 0; i < 8; ++i) {
    a0[i] = c0[i]; a0[i + 8] = c1[i];
    a1[i] = d0[i]; a1[i + 8] = d1[i];
  }
#pragma unroll
  for (int t = 0; t < 4; ++t)
    b[t] = *(const v16bf*)(pb + (long long)t * 16 * KDIM + OFFE);
}

__device__ __forceinline__ void mma_set(const v16bf& a0, const v16bf& a1,
                                        const v16bf (&b)[4], v8f (&acc)[2][4]) {
#pragma unroll
  for (int t = 0; t < 4; ++t) {
    acc[0][t] = __builtin_amdgcn_wmma_f32_16x16x32_bf16(
        false, a0, false, b[t], (short)0, acc[0][t], false, false);
    acc[1][t] = __builtin_amdgcn_wmma_f32_16x16x32_bf16(
        false, a1, false, b[t], (short)0, acc[1][t], false, false);
  }
}

// ---- Kernel 3: y = Xb @ Wb^T + bias ---------------------------------------
// 256 threads = 8 waves (2M x 4N). Wave tile 32Mx64N -> 8 accumulators.
__global__ __launch_bounds__(256) void ggml_wmma_gemm(
    const unsigned short* __restrict__ Xb_, const unsigned short* __restrict__ Wb_,
    const float* __restrict__ bias, float* __restrict__ Y, int O) {
  const __bf16* __restrict__ Xb = reinterpret_cast<const __bf16*>(Xb_);
  const __bf16* __restrict__ Wb = reinterpret_cast<const __bf16*>(Wb_);

  const int lane = threadIdx.x & 31;
  const int wave = threadIdx.x >> 5;
  const int half = lane >> 4;     // 0: K-low half, 1: K-high half
  const int lr   = lane & 15;
  const int mw   = wave >> 2;     // 0..1  -> 32-row M sub-tile
  const int nw   = wave & 3;      // 0..3  -> 64-col N sub-tile
  const int M0   = blockIdx.x * 64 + mw * 32;
  const int N0   = blockIdx.y * 256 + nw * 64;

  v8f acc[2][4];
#pragma unroll
  for (int mi = 0; mi < 2; ++mi)
#pragma unroll
    for (int t = 0; t < 4; ++t)
      acc[mi][t] = (v8f){0.f,0.f,0.f,0.f,0.f,0.f,0.f,0.f};

  // Two per-lane streaming pointers; everything else is an immediate offset.
  const __bf16* pa = Xb + (long long)(M0 + lr) * KDIM + half * 16;
  const __bf16* pb = Wb + (long long)(N0 + lr) * KDIM + half * 16;

  // Ping-pong fragment buffers (no rotation copies anywhere).
  v16bf a00, a01, b0[4];   // K-step "even"  (k)
  v16bf a10, a11, b1[4];   // K-step "odd"   (k+32)

  // Prologue: fill both buffers for k=0 and k=32.
  load_set<0>(pa, pb, a00, a01, b0);
  load_set<32>(pa, pb, a10, a11, b1);
  pa += 64; pb += 64;

  // Steady state: consume a buffer, immediately refill it for k (pointer base).
#pragma unroll 1
  for (int k = 64; k <= KDIM - 64; k += 64) {
    __builtin_prefetch((const void*)(pa + 256), 0, 3);
    __builtin_prefetch((const void*)(pb + 256), 0, 3);
    mma_set(a00, a01, b0, acc);          // consume k-64
    load_set<0>(pa, pb, a00, a01, b0);   // refill with k
    mma_set(a10, a11, b1, acc);          // consume k-32
    load_set<32>(pa, pb, a10, a11, b1);  // refill with k+32
    pa += 64; pb += 64;
  }

  // Drain the last two K-steps.
  mma_set(a00, a01, b0, acc);
  mma_set(a10, a11, b1, acc);

  // Store: C/D map -> VGPR j holds (M = M0 + mi*16 + half*8 + j, N = N0+t*16+lr)
#pragma unroll
  for (int mi = 0; mi < 2; ++mi) {
    const int row0 = M0 + mi * 16 + half * 8;
#pragma unroll
    for (int t = 0; t < 4; ++t) {
      const int col = N0 + t * 16 + lr;
      const float bv = bias[col];
#pragma unroll
      for (int j = 0; j < 8; ++j) {
        Y[(long long)(row0 + j) * O + col] = acc[mi][t][j] + bv;
      }
    }
  }
}

// ---------------------------------------------------------------------------
extern "C" void kernel_launch(void* const* d_in, const int* in_sizes, int n_in,
                              void* d_out, int out_size, void* d_ws, size_t ws_size,
                              hipStream_t stream) {
  const float* x      = (const float*)d_in[0];
  const int*   qw     = (const int*)d_in[1];
  const float* scales = (const float*)d_in[2];
  const float* la     = (const float*)d_in[3];
  const float* lb     = (const float*)d_in[4];
  const float* bias   = (const float*)d_in[5];
  float* y = (float*)d_out;

  const int O = in_sizes[5];                       // 4096
  const int I = (int)((long long)in_sizes[1] / O); // 4096 (== KDIM)
  const int T = (int)((long long)in_sizes[0] / I); // 8192
  const int R = in_sizes[3] / O;                   // 16

  // Workspace: Xb [T*I] bf16, then Wb [O*I] bf16.
  unsigned short* Xb = (unsigned short*)d_ws;
  unsigned short* Wb = (unsigned short*)((char*)d_ws + (size_t)T * I * sizeof(unsigned short));

  {
    long long n = (long long)T * I;
    unsigned int blocks = (unsigned int)((n + 255) / 256);
    ggml_prep_x<<<dim3(blocks), dim3(256), 0, stream>>>(x, Xb, n);
  }
  {
    long long n = (long long)O * I;
    unsigned int blocks = (unsigned int)((n + 255) / 256);
    ggml_prep_w<<<dim3(blocks), dim3(256), 0, stream>>>(qw, scales, la, lb, Wb, O, I, R);
  }
  {
    dim3 grid((unsigned)(T / 64), (unsigned)(O / 256));
    ggml_wmma_gemm<<<grid, dim3(256), 0, stream>>>(Xb, Wb, bias, y, O);
  }
}
